// HTR_28329604285247
// MI455X (gfx1250) — matrix operand order repaired and checked by reading, not compiled, not run on hardware
//
#include <hip/hip_runtime.h>
#include <hip/hip_bf16.h>

typedef __bf16 bf16_t;
typedef __attribute__((ext_vector_type(8)))  bf16_t v8bf;
typedef __attribute__((ext_vector_type(16))) bf16_t v16bf;
typedef __attribute__((ext_vector_type(8)))  float  v8f;

__device__ __forceinline__ v8f wmma_bf16(v16bf a, v16bf b, v8f c) {
    // D = A(16x32 bf16) x B(32x16 bf16) + C(16x16 f32)
    return __builtin_amdgcn_wmma_f32_16x16x32_bf16(
        /*neg_a=*/false, a, /*neg_b=*/false, b,
        /*c_mod=*/(short)0, c, /*reuse_a=*/false, /*reuse_b=*/false);
}

__device__ __forceinline__ v16bf cat8(v8bf lo, v8bf hi) {
    return __builtin_shufflevector(lo, hi, 0,1,2,3,4,5,6,7,8,9,10,11,12,13,14,15);
}

// B fragment: lane n = lane&15 (+16*nt applied by caller), K values k0..k0+7 and k0+16..k0+23.
__device__ __forceinline__ v16bf load_bfrag(const bf16_t* W, int ldk, int n, int k0) {
    const v8bf* p = (const v8bf*)(W + (size_t)n * ldk + k0);
    return cat8(p[0], p[2]);
}

// A fragment from an fp32 row: elements 0..7 = row[k0..k0+7], 8..15 = row[k0+16..k0+23]
__device__ __forceinline__ v16bf afrag_f32(const float* row, int k0) {
    v16bf a;
#pragma unroll
    for (int i = 0; i < 8; ++i) {
        a[i]     = (bf16_t)row[k0 + i];
        a[8 + i] = (bf16_t)row[k0 + 16 + i];
    }
    return a;
}

// A fragment from per-wave LDS bf16 buffer (row-major, stride ldk)
__device__ __forceinline__ v16bf afrag_lds(const bf16_t* hb, int ldk, int m, int k0) {
    const v8bf* p0 = (const v8bf*)(hb + m * ldk + k0);
    const v8bf* p1 = (const v8bf*)(hb + m * ldk + k0 + 16);
    return cat8(*p0, *p1);
}

__device__ __forceinline__ v8f splat8(float x) {
    v8f v;
#pragma unroll
    for (int i = 0; i < 8; ++i) v[i] = x;
    return v;
}

__device__ __forceinline__ float silu(float x) {
    return x / (1.0f + __expf(-x));
}

__device__ __forceinline__ void wave_sync() {
    __builtin_amdgcn_fence(__ATOMIC_RELEASE, "wavefront");
    __builtin_amdgcn_wave_barrier();
    __builtin_amdgcn_fence(__ATOMIC_ACQUIRE, "wavefront");
}

// ---------------------------------------------------------------------------
// Kernel 0: convert all weight matrices fp32 -> bf16 into workspace
// layout (elements): Wvq[8192] Wk1[8192] Wk2[8192] gw_w1[4096] gw_w2[8192]
//                    gt_w1[16384] gt_w2[16384]   total 69632
// ---------------------------------------------------------------------------
__global__ void cvt_weights(const float* wq, const float* wk1, const float* wk2,
                            const float* g1, const float* g2,
                            const float* t1, const float* t2, bf16_t* dst) {
    int i = blockIdx.x * blockDim.x + threadIdx.x;
    if (i >= 69632) return;
    float v;
    if      (i < 8192)  v = wq[i];
    else if (i < 16384) v = wk1[i - 8192];
    else if (i < 24576) v = wk2[i - 16384];
    else if (i < 28672) v = g1[i - 24576];
    else if (i < 36864) v = g2[i - 28672];
    else if (i < 53248) v = t1[i - 36864];
    else                v = t2[i - 53248];
    dst[i] = (bf16_t)v;
}

// ---------------------------------------------------------------------------
// Kernel 1: node projections.  One wave = 2 nodes = 16 rows of X_cat.
// Qall[n][m][o] = X_cat[n][m] . Wvq[o]      (m = 0..7, o = 0..63)
// Kall[n][m][o] = X1[n][m] . Wk1[o] (m<3) else X2[n][m-3] . Wk2[o]
// ---------------------------------------------------------------------------
__global__ void proj_kernel(const float* __restrict__ X1, const float* __restrict__ X2,
                            const bf16_t* __restrict__ wq, const bf16_t* __restrict__ wk1,
                            const bf16_t* __restrict__ wk2,
                            float* __restrict__ Qall, float* __restrict__ Kall, int Nn) {
    const int wid  = threadIdx.x >> 5;
    const int lane = threadIdx.x & 31;
    const int tile = blockIdx.x * (blockDim.x >> 5) + wid;
    const int ntiles = (Nn + 1) >> 1;
    if (tile >= ntiles) return;                      // wave-uniform

    const int Mrow  = lane & 15;                     // A row / B col / D col
    const int hi    = lane >> 4;                     // lane half
    const int kbase = hi ? 8 : 0;

    // A row source: node/m for this lane's A-matrix row
    int node_a = tile * 2 + (Mrow >> 3);
    if (node_a >= Nn) node_a = Nn - 1;
    const int m = Mrow & 7;
    const float* arow = (m < 3) ? (X1 + ((size_t)node_a * 3 + m) * 128)
                                : (X2 + ((size_t)node_a * 5 + (m - 3)) * 128);

    v8f aq[4], ak1[4], ak2[4];
#pragma unroll
    for (int nt = 0; nt < 4; ++nt) { aq[nt] = splat8(0.f); ak1[nt] = splat8(0.f); ak2[nt] = splat8(0.f); }

#pragma unroll
    for (int kc = 0; kc < 4; ++kc) {                 // K = 128 in 4 chunks of 32
        const v16bf A  = afrag_f32(arow, kbase + 32 * kc);
        const int   kb = hi * 8 + 32 * kc;           // B-frag K start
#pragma unroll
        for (int nt = 0; nt < 4; ++nt) {             // 64 outputs = 4 N-tiles
            const int n = nt * 16 + Mrow;
            aq[nt]  = wmma_bf16(A, load_bfrag(wq,  128, n, kb), aq[nt]);
            ak1[nt] = wmma_bf16(A, load_bfrag(wk1, 128, n, kb), ak1[nt]);
            ak2[nt] = wmma_bf16(A, load_bfrag(wk2, 128, n, kb), ak2[nt]);
        }
    }

    // D layout: element v of this lane = (row M = v + 8*hi, col = nt*16 + Mrow)
    const int noded = tile * 2 + hi;
    if (noded < Nn) {
        const size_t base = (size_t)noded * 512;     // 8*64
#pragma unroll
        for (int nt = 0; nt < 4; ++nt) {
#pragma unroll
            for (int v = 0; v < 8; ++v) {
                const size_t idx = base + (size_t)v * 64 + nt * 16 + Mrow;
                Qall[idx] = aq[nt][v];
                Kall[idx] = (v < 3) ? ak1[nt][v] : ak2[nt][v];   // m = v
            }
        }
    }
}

// ---------------------------------------------------------------------------
// Per-lane: 8 consecutive w features starting at o=ob for edge (Qp,Kp,r1,r2).
// Uses Q'.K' = Q.K - (Q.r)(K.r)(2-|r|^2) per rejection group.
// ---------------------------------------------------------------------------
__device__ __forceinline__ void w_chunk(const float* __restrict__ Qp,
                                        const float* __restrict__ Kp, int ob,
                                        const float r1[3], const float r2[5],
                                        float c1, float c2, float* __restrict__ wout) {
    float s[8], dq1[8], dk1[8], dq2[8], dk2[8];
#pragma unroll
    for (int i = 0; i < 8; ++i) { s[i] = 0.f; dq1[i] = 0.f; dk1[i] = 0.f; dq2[i] = 0.f; dk2[i] = 0.f; }
#pragma unroll
    for (int mm = 0; mm < 8; ++mm) {
        const float4* qp = (const float4*)(Qp + mm * 64 + ob);
        const float4* kp = (const float4*)(Kp + mm * 64 + ob);
        const float4 qa = qp[0], qb = qp[1], ka = kp[0], kb = kp[1];
        const float qv[8] = {qa.x, qa.y, qa.z, qa.w, qb.x, qb.y, qb.z, qb.w};
        const float kv[8] = {ka.x, ka.y, ka.z, ka.w, kb.x, kb.y, kb.z, kb.w};
        const float r = (mm < 3) ? r1[mm] : r2[mm - 3];
#pragma unroll
        for (int i = 0; i < 8; ++i) {
            s[i] += qv[i] * kv[i];
            if (mm < 3) { dq1[i] += qv[i] * r; dk1[i] += kv[i] * r; }
            else        { dq2[i] += qv[i] * r; dk2[i] += kv[i] * r; }
        }
    }
#pragma unroll
    for (int i = 0; i < 8; ++i)
        wout[i] = s[i] - dq1[i] * dk1[i] * c1 - dq2[i] * dk2[i] * c2;
}

// ---------------------------------------------------------------------------
// Kernel 2: fused edge gather + rejection + w + both MLPs + combine.
// One wave = 16 edges.  out = t + mlp_w(w) * mlp_t(t)
// ---------------------------------------------------------------------------
__global__ void edge_mlp_kernel(const float* __restrict__ t, const int* __restrict__ edge,
                                const float* __restrict__ rt1, const float* __restrict__ rt2,
                                const float* __restrict__ Qall, const float* __restrict__ Kall,
                                const bf16_t* __restrict__ gw1, const float* __restrict__ gwb1,
                                const bf16_t* __restrict__ gw2, const float* __restrict__ gwb2,
                                const bf16_t* __restrict__ gt1, const float* __restrict__ gtb1,
                                const bf16_t* __restrict__ gt2, const float* __restrict__ gtb2,
                                float* __restrict__ out, int E) {
    __shared__ __align__(16) bf16_t hbuf[8][16 * 128];   // per-wave relayout scratch (4 KB each)

    const int wid  = threadIdx.x >> 5;
    const int lane = threadIdx.x & 31;
    const int tile = blockIdx.x * (blockDim.x >> 5) + wid;
    if (tile * 16 >= E) return;                           // wave-uniform

    const int Mrow  = lane & 15;
    const int hi    = lane >> 4;
    const int kbase = hi ? 8 : 0;
    bf16_t* hb = hbuf[wid];

    // ----- per-lane edge for the A-matrix row it owns -----
    int e = tile * 16 + Mrow;
    if (e >= E) e = E - 1;                                // safe duplicate (gather only)
    const int i0 = edge[2 * (size_t)e + 0];
    const int j0 = edge[2 * (size_t)e + 1];
    float r1[3], r2[5];
#pragma unroll
    for (int k = 0; k < 3; ++k) r1[k] = rt1[(size_t)e * 3 + k];
#pragma unroll
    for (int k = 0; k < 5; ++k) r2[k] = rt2[(size_t)e * 5 + k];
    const float c1 = 2.f - (r1[0]*r1[0] + r1[1]*r1[1] + r1[2]*r1[2]);
    const float c2 = 2.f - (r2[0]*r2[0] + r2[1]*r2[1] + r2[2]*r2[2] + r2[3]*r2[3] + r2[4]*r2[4]);
    const float* Qp = Qall + (size_t)i0 * 512;
    const float* Kp = Kall + (size_t)j0 * 512;

    // ----- build w A-fragments directly (only the 32 features this lane needs) -----
    v16bf fragW[2];
#pragma unroll
    for (int kc = 0; kc < 2; ++kc) {
        float wlo[8], whi[8];
        w_chunk(Qp, Kp, 32 * kc + kbase,      r1, r2, c1, c2, wlo);
        w_chunk(Qp, Kp, 32 * kc + kbase + 16, r1, r2, c1, c2, whi);
#pragma unroll
        for (int i = 0; i < 8; ++i) { fragW[kc][i] = (bf16_t)wlo[i]; fragW[kc][8 + i] = (bf16_t)whi[i]; }
    }

    // ----- GEMM1: h1 = silu(w @ gw_w1^T + b1)   (16x64, K=64) -----
    v8f h1[4];
#pragma unroll
    for (int nt = 0; nt < 4; ++nt) h1[nt] = splat8(gwb1[nt * 16 + Mrow]);
#pragma unroll
    for (int kc = 0; kc < 2; ++kc) {
        const int kb = hi * 8 + 32 * kc;
#pragma unroll
        for (int nt = 0; nt < 4; ++nt)
            h1[nt] = wmma_bf16(fragW[kc], load_bfrag(gw1, 64, nt * 16 + Mrow, kb), h1[nt]);
    }
    // silu -> LDS (D layout -> row-major bf16, stride 64)
#pragma unroll
    for (int nt = 0; nt < 4; ++nt)
#pragma unroll
        for (int v = 0; v < 8; ++v)
            hb[(v + 8 * hi) * 64 + nt * 16 + Mrow] = (bf16_t)silu(h1[nt][v]);
    wave_sync();

    // ----- GEMM2: y1 = h1 @ gw_w2^T + b2   (16x128, K=64) -----
    v8f y1[8];
#pragma unroll
    for (int nt = 0; nt < 8; ++nt) y1[nt] = splat8(gwb2[nt * 16 + Mrow]);
#pragma unroll
    for (int kc = 0; kc < 2; ++kc) {
        const v16bf A  = afrag_lds(hb, 64, Mrow, kbase + 32 * kc);
        const int   kb = hi * 8 + 32 * kc;
#pragma unroll
        for (int nt = 0; nt < 8; ++nt)
            y1[nt] = wmma_bf16(A, load_bfrag(gw2, 64, nt * 16 + Mrow, kb), y1[nt]);
    }
    wave_sync();   // all fragW/h1 LDS reads done before hb is overwritten below

    // ----- GEMM3: h2 = silu(t @ gt_w1^T + b1)   (16x128, K=128) -----
    v8f h2[8];
#pragma unroll
    for (int nt = 0; nt < 8; ++nt) h2[nt] = splat8(gtb1[nt * 16 + Mrow]);
    const float* trow = t + (size_t)e * 128;
#pragma unroll
    for (int kc = 0; kc < 4; ++kc) {
        const v16bf A  = afrag_f32(trow, kbase + 32 * kc);
        const int   kb = hi * 8 + 32 * kc;
#pragma unroll
        for (int nt = 0; nt < 8; ++nt)
            h2[nt] = wmma_bf16(A, load_bfrag(gt1, 128, nt * 16 + Mrow, kb), h2[nt]);
    }
    // silu -> LDS (stride 128)
#pragma unroll
    for (int nt = 0; nt < 8; ++nt)
#pragma unroll
        for (int v = 0; v < 8; ++v)
            hb[(v + 8 * hi) * 128 + nt * 16 + Mrow] = (bf16_t)silu(h2[nt][v]);
    wave_sync();

    // ----- GEMM4: y2 = h2 @ gt_w2^T + b2   (16x128, K=128) -----
    v8f y2[8];
#pragma unroll
    for (int nt = 0; nt < 8; ++nt) y2[nt] = splat8(gtb2[nt * 16 + Mrow]);
#pragma unroll
    for (int kc = 0; kc < 4; ++kc) {
        const v16bf A  = afrag_lds(hb, 128, Mrow, kbase + 32 * kc);
        const int   kb = hi * 8 + 32 * kc;
#pragma unroll
        for (int nt = 0; nt < 8; ++nt)
            y2[nt] = wmma_bf16(A, load_bfrag(gt2, 128, nt * 16 + Mrow, kb), y2[nt]);
    }

    // ----- combine: out = t + y1 * y2   (D layout scatter) -----
#pragma unroll
    for (int nt = 0; nt < 8; ++nt) {
#pragma unroll
        for (int v = 0; v < 8; ++v) {
            const int ee = tile * 16 + v + 8 * hi;
            if (ee < E) {
                const size_t idx = (size_t)ee * 128 + nt * 16 + Mrow;
                out[idx] = t[idx] + y1[nt][v] * y2[nt][v];
            }
        }
    }
}

// ---------------------------------------------------------------------------
extern "C" void kernel_launch(void* const* d_in, const int* in_sizes, int n_in,
                              void* d_out, int out_size, void* d_ws, size_t ws_size,
                              hipStream_t stream) {
    const float* t    = (const float*)d_in[0];
    const float* X1   = (const float*)d_in[1];
    const float* X2   = (const float*)d_in[2];
    const int*   edge = (const int*)  d_in[3];
    const float* rt1  = (const float*)d_in[4];
    const float* rt2  = (const float*)d_in[5];
    const float* Wvq  = (const float*)d_in[6];
    const float* Wk1  = (const float*)d_in[7];
    const float* Wk2  = (const float*)d_in[8];
    const float* gw_w1 = (const float*)d_in[9];
    const float* gw_b1 = (const float*)d_in[10];
    const float* gw_w2 = (const float*)d_in[11];
    const float* gw_b2 = (const float*)d_in[12];
    const float* gt_w1 = (const float*)d_in[13];
    const float* gt_b1 = (const float*)d_in[14];
    const float* gt_w2 = (const float*)d_in[15];
    const float* gt_b2 = (const float*)d_in[16];
    float* out = (float*)d_out;

    const int E = in_sizes[0] / 128;        // t is (E,128)
    const int N = in_sizes[1] / (3 * 128);  // X1 is (N,3,128)

    // workspace layout
    float*  Qall = (float*)d_ws;
    float*  Kall = Qall + (size_t)N * 512;
    bf16_t* wb   = (bf16_t*)(Kall + (size_t)N * 512);
    bf16_t* wq  = wb;
    bf16_t* wk1 = wb + 8192;
    bf16_t* wk2 = wb + 16384;
    bf16_t* gw1 = wb + 24576;
    bf16_t* gw2 = wb + 28672;
    bf16_t* gt1 = wb + 36864;
    bf16_t* gt2 = wb + 53248;

    cvt_weights<<<(69632 + 255) / 256, 256, 0, stream>>>(Wvq, Wk1, Wk2, gw_w1, gw_w2, gt_w1, gt_w2, wb);

    const int ptiles  = (N + 1) / 2;
    const int pblocks = (ptiles + 7) / 8;
    proj_kernel<<<pblocks, 256, 0, stream>>>(X1, X2, wq, wk1, wk2, Qall, Kall, N);

    const int etiles  = (E + 15) / 16;
    const int eblocks = (etiles + 7) / 8;
    edge_mlp_kernel<<<eblocks, 256, 0, stream>>>(t, edge, rt1, rt2, Qall, Kall,
                                                 gw1, gw_b1, gw2, gw_b2,
                                                 gt1, gt_b1, gt2, gt_b2, out, E);
}
